// TinyMoEModel_62466004353266
// MI455X (gfx1250) — matrix operand order repaired and checked by reading, not compiled, not run on hardware
//
#include <hip/hip_runtime.h>
#include <hip/hip_bf16.h>
#include <math.h>

// Problem constants (reference: B=8, S=2048, H=1024, E=8, CAP_FACTOR=1.25)
#define TOK 16384           // B*S tokens
#define HD  1024            // hidden
#define NE  8               // experts
#define CAP 2560            // ceil(1.25 * TOK / NE)

typedef __bf16 bf16_t;
typedef __bf16 v16bf __attribute__((ext_vector_type(16)));
typedef __bf16 v8bf  __attribute__((ext_vector_type(8)));
typedef float  v8f   __attribute__((ext_vector_type(8)));

// gfx1250 async global->LDS path (ASYNCcnt-tracked DMA, no VGPR round-trip)
#if defined(__has_builtin)
#  if __has_builtin(__builtin_amdgcn_global_load_async_to_lds_b128)
#    define HAS_ASYNC_LDS 1
#  endif
#endif
#ifndef HAS_ASYNC_LDS
#  define HAS_ASYNC_LDS 0
#endif

// Exact pointee type from the builtin's signature: gcc-vector int4,
// global (AS1) source and LDS (AS3) destination pointers.
typedef int v4i_t __attribute__((vector_size(4 * sizeof(int))));
typedef __attribute__((address_space(1))) v4i_t* gptr_v4i;
typedef __attribute__((address_space(3))) v4i_t* lptr_v4i;

__device__ __forceinline__ void async_copy16(const bf16_t* g, bf16_t* l) {
#if HAS_ASYNC_LDS
  __builtin_amdgcn_global_load_async_to_lds_b128(
      (gptr_v4i)(void*)const_cast<bf16_t*>(g), (lptr_v4i)(void*)l, 0, 0);
#else
  *(uint4*)l = *(const uint4*)g;
#endif
}

__device__ __forceinline__ void wait_async0() {
#if HAS_ASYNC_LDS
#  if __has_builtin(__builtin_amdgcn_s_wait_asynccnt)
  __builtin_amdgcn_s_wait_asynccnt(0);
#  else
  asm volatile("s_wait_asynccnt 0x0" ::: "memory");
#  endif
#endif
}

__device__ __forceinline__ v16bf cat16(v8bf lo, v8bf hi) {
  v16bf r;
#pragma unroll
  for (int i = 0; i < 8; ++i) { r[i] = lo[i]; r[i + 8] = hi[i]; }
  return r;
}

// ---------------------------------------------------------------------------
// Tiled bf16 WMMA GEMM:  C[M,N] = A[M,K] @ B[K,N] (+bias), B given transposed
// (BT is [N,K] row-major so both operands are contiguous along K).
// Block tile 128x128x32, 256 threads = 8 waves, each wave owns a 32x64 tile
// (2x4 grid of 16x16 WMMA tiles), f32 accumulate. Global->LDS staging uses
// double-buffered GLOBAL_LOAD_ASYNC_TO_LDS_B128 overlapped with WMMA compute.
// EPI: 0 = f32 out, 1 = bf16 out, 2 = relu -> bf16 out. Batched via blockIdx.z.
// ---------------------------------------------------------------------------
#define BM 128
#define BN 128
#define BK 32
#define LDT 40  // padded LDS row stride (bf16): 80B, multiple of 16B

template <int EPI>
__global__ __launch_bounds__(256)
void gemm_bf16(const bf16_t* __restrict__ A, const bf16_t* __restrict__ BT,
               const float* __restrict__ bias,
               float* __restrict__ outF, bf16_t* __restrict__ outB,
               int N, int K,
               long long sA, long long sB, long long sBias, long long sOut) {
  __shared__ bf16_t As[2][BM * LDT];
  __shared__ bf16_t Bs[2][BN * LDT];

  const int bz = blockIdx.z;
  A  += (long long)bz * sA;
  BT += (long long)bz * sB;
  const float* biasz = bias ? bias + (long long)bz * sBias : nullptr;

  const int m0 = blockIdx.y * BM;
  const int n0 = blockIdx.x * BN;

  const int tid  = threadIdx.x;
  const int wave = tid >> 5;
  const int lane = tid & 31;
  const int wm   = wave & 3;   // 4 waves along M
  const int wn   = wave >> 2;  // 2 waves along N
  const int lm   = lane & 15;
  const int lh   = lane >> 4;

  // This thread's two staging chunks (8 bf16 = 16B each) per matrix per tile.
  const int row0 = (tid + 0)   >> 2, c80 = ((tid + 0)   & 3) << 3;
  const int row1 = (tid + 256) >> 2, c81 = ((tid + 256) & 3) << 3;

  v8f acc[2][4];
#pragma unroll
  for (int i = 0; i < 2; ++i)
#pragma unroll
    for (int j = 0; j < 4; ++j)
#pragma unroll
      for (int k = 0; k < 8; ++k) acc[i][j][k] = 0.f;

  auto stage = [&](int buf, int kt) {
    const int k0 = kt * BK;
    async_copy16(A  + (long long)(m0 + row0) * K + k0 + c80, &As[buf][row0 * LDT + c80]);
    async_copy16(BT + (long long)(n0 + row0) * K + k0 + c80, &Bs[buf][row0 * LDT + c80]);
    async_copy16(A  + (long long)(m0 + row1) * K + k0 + c81, &As[buf][row1 * LDT + c81]);
    async_copy16(BT + (long long)(n0 + row1) * K + k0 + c81, &Bs[buf][row1 * LDT + c81]);
  };

  const int ktiles = K / BK;
  stage(0, 0);

  int buf = 0;
  for (int kt = 0; kt < ktiles; ++kt) {
    wait_async0();        // this wave's async DMAs for `buf` are in LDS
    __syncthreads();      // everyone's are
    if (kt + 1 < ktiles) stage(buf ^ 1, kt + 1);  // overlap next tile's DMA

    const bf16_t* Asb = As[buf];
    const bf16_t* Bsb = Bs[buf];

    // A fragments (ISA 16-bit A 16x32 layout): lanes 0-15 -> K {0..7,16..23},
    // lanes 16-31 -> K {8..15,24..31}, M = lane%16.
    v16bf afrag[2];
#pragma unroll
    for (int sm = 0; sm < 2; ++sm) {
      const bf16_t* ap = Asb + (wm * 32 + sm * 16 + lm) * LDT;
      afrag[sm] = cat16(*(const v8bf*)(ap + lh * 8),
                        *(const v8bf*)(ap + 16 + lh * 8));
    }
#pragma unroll
    for (int sn = 0; sn < 4; ++sn) {
      // B fragment: N = lane%16, K same as A: lanes 0-15 {0..7,16..23}, 16-31 {8..15,24..31}
      const bf16_t* bp = Bsb + (wn * 64 + sn * 16 + lm) * LDT + lh * 8;
      const v16bf bfrag = cat16(*(const v8bf*)bp, *(const v8bf*)(bp + 16));
#pragma unroll
      for (int sm = 0; sm < 2; ++sm) {
        acc[sm][sn] = __builtin_amdgcn_wmma_f32_16x16x32_bf16(
            false, afrag[sm], false, bfrag, (short)0, acc[sm][sn], false, false);
      }
    }
    __syncthreads();      // all reads of `buf` retired before it is re-filled
    buf ^= 1;
  }

  // Epilogue. C/D layout: lane n=lane%16, rows m = (lane/16)*8 + v.
#pragma unroll
  for (int sm = 0; sm < 2; ++sm) {
#pragma unroll
    for (int sn = 0; sn < 4; ++sn) {
      const int n  = n0 + wn * 64 + sn * 16 + lm;
      const int mb = m0 + wm * 32 + sm * 16 + lh * 8;
      const float bn = biasz ? biasz[n] : 0.f;
#pragma unroll
      for (int v = 0; v < 8; ++v) {
        float val = acc[sm][sn][v] + bn;
        if (EPI == 2) val = fmaxf(val, 0.f);
        const long long idx = (long long)bz * sOut + (long long)(mb + v) * N + n;
        if (EPI == 0) outF[idx] = val;
        else          outB[idx] = (bf16_t)val;
      }
    }
  }
}

// ---------------------------------------------------------------------------
// Helper kernels
// ---------------------------------------------------------------------------
__global__ void cvt_f32_to_bf16(const float* __restrict__ in,
                                bf16_t* __restrict__ out, long long n) {
  long long i = (long long)blockIdx.x * blockDim.x + threadIdx.x;
  const long long stride = (long long)gridDim.x * blockDim.x;
  for (; i < n; i += stride) out[i] = (bf16_t)in[i];
}

// W [K,N] f32 (row-major) -> WT [N,K] bf16 (row-major), batched over z.
__global__ void transpose_cvt(const float* __restrict__ W, bf16_t* __restrict__ WT,
                              int K, int N, long long sIn, long long sOut) {
  __shared__ float tile[32][33];
  const int bz = blockIdx.z;
  const float* Wz = W + (long long)bz * sIn;
  bf16_t* WTz = WT + (long long)bz * sOut;
  const int n0 = blockIdx.x * 32, k0 = blockIdx.y * 32;
  const int tx = threadIdx.x, ty = threadIdx.y;  // (32, 8)
  for (int i = ty; i < 32; i += 8)
    tile[i][tx] = Wz[(long long)(k0 + i) * N + n0 + tx];
  __syncthreads();
  for (int i = ty; i < 32; i += 8)
    WTz[(long long)(n0 + i) * K + k0 + tx] = (bf16_t)tile[tx][i];
}

__global__ void fill_zero_u4(uint4* __restrict__ p, long long nchunks) {
  long long i = (long long)blockIdx.x * blockDim.x + threadIdx.x;
  const long long stride = (long long)gridDim.x * blockDim.x;
  const uint4 z = {0u, 0u, 0u, 0u};
  for (; i < nchunks; i += stride) p[i] = z;
}

// Per-token: gate logits (H x 8), 2-way coef logits (H x 2), softmax, argmax.
__global__ __launch_bounds__(128)
void gate_coef(const bf16_t* __restrict__ t, const float* __restrict__ Wg,
               const float* __restrict__ Wc, const float* __restrict__ bc,
               int* __restrict__ eidx, float* __restrict__ gmax,
               float* __restrict__ coef0, float* __restrict__ coef1) {
  const int tok = blockIdx.x;
  const int tid = threadIdx.x;
  float g[8] = {0, 0, 0, 0, 0, 0, 0, 0};
  float c[2] = {0, 0};
  const bf16_t* tr = t + (long long)tok * HD;
  for (int h = tid; h < HD; h += 128) {
    const float xv = (float)tr[h];
    const float* wg = Wg + h * NE;
#pragma unroll
    for (int j = 0; j < 8; ++j) g[j] += xv * wg[j];
    const float* wc = Wc + h * 2;
    c[0] += xv * wc[0];
    c[1] += xv * wc[1];
  }
#pragma unroll
  for (int j = 0; j < 8; ++j)
    for (int off = 16; off; off >>= 1) g[j] += __shfl_xor(g[j], off, 32);
#pragma unroll
  for (int j = 0; j < 2; ++j)
    for (int off = 16; off; off >>= 1) c[j] += __shfl_xor(c[j], off, 32);

  __shared__ float red[4][10];
  const int wave = tid >> 5, lane = tid & 31;
  if (lane == 0) {
#pragma unroll
    for (int j = 0; j < 8; ++j) red[wave][j] = g[j];
    red[wave][8] = c[0];
    red[wave][9] = c[1];
  }
  __syncthreads();
  if (tid == 0) {
    float gg[8], cc[2];
#pragma unroll
    for (int j = 0; j < 8; ++j)
      gg[j] = red[0][j] + red[1][j] + red[2][j] + red[3][j];
    cc[0] = red[0][8] + red[1][8] + red[2][8] + red[3][8];
    cc[1] = red[0][9] + red[1][9] + red[2][9] + red[3][9];
    int bi = 0;
    float bm = gg[0];
    for (int j = 1; j < 8; ++j)
      if (gg[j] > bm) { bm = gg[j]; bi = j; }   // first-max, matches jnp.argmax
    float den = 0.f;
    for (int j = 0; j < 8; ++j) den += __expf(gg[j] - bm);
    eidx[tok] = bi;
    gmax[tok] = 1.f / den;                      // softmax value at argmax
    const float c0 = cc[0] + bc[0], c1 = cc[1] + bc[1];
    const float cm = fmaxf(c0, c1);
    const float e0 = __expf(c0 - cm), e1 = __expf(c1 - cm);
    coef0[tok] = e0 / (e0 + e1);
    coef1[tok] = e1 / (e0 + e1);
  }
}

// Ordered per-expert prefix scan over tokens (reference cumsum semantics).
// Single wave32; wave-wide ranks via ballot, running bases in LDS.
__global__ void scan_assign(const int* __restrict__ eidx, const float* __restrict__ gmax,
                            int* __restrict__ slot, int* __restrict__ keep,
                            float* __restrict__ gv) {
  __shared__ int base[NE];
  const int lane = threadIdx.x;
  if (lane < NE) base[lane] = 0;
  __syncthreads();
  const unsigned lt = (1u << lane) - 1u;
  for (int cch = 0; cch < TOK / 32; ++cch) {
    const int t = cch * 32 + lane;
    const int e = eidx[t];
    int rank = 0, mycnt = 0;
#pragma unroll
    for (int ex = 0; ex < NE; ++ex) {
      const unsigned m = __builtin_amdgcn_ballot_w32(e == ex);
      if (e == ex) rank = __popc(m & lt);
      if (lane == ex) mycnt = __popc(m);
    }
    const int pos = base[e] + rank;
    __syncthreads();
    if (lane < NE) base[lane] += mycnt;
    __syncthreads();
    const int kp = pos < CAP;
    slot[t] = kp ? pos : (CAP - 1);
    keep[t] = kp;
    gv[t]   = kp ? gmax[t] : 0.f;
  }
}

// Scatter kept token rows into [E, CAP, H] dispatch buffer.
__global__ __launch_bounds__(128)
void dispatch_rows(const bf16_t* __restrict__ t, const int* __restrict__ eidx,
                   const int* __restrict__ slot, const int* __restrict__ keep,
                   bf16_t* __restrict__ disp) {
  const int tok = blockIdx.x;
  if (!keep[tok]) return;
  const long long dst = ((long long)eidx[tok] * CAP + slot[tok]) * HD;
  const uint4* src = (const uint4*)(t + (long long)tok * HD);
  uint4* d = (uint4*)(disp + dst);
  d[threadIdx.x] = src[threadIdx.x];  // 128 threads x 8 bf16 = 1024
}

// mix = eo[eidx,slot] * gv * coef0 + res * coef1  (bf16 out for final GEMM)
__global__ void mix_kernel(const bf16_t* __restrict__ eo, const bf16_t* __restrict__ res,
                           const int* __restrict__ eidx, const int* __restrict__ slot,
                           const float* __restrict__ gv, const float* __restrict__ coef0,
                           const float* __restrict__ coef1, bf16_t* __restrict__ mix) {
  const long long nch = (long long)TOK * HD / 8;
  long long i = (long long)blockIdx.x * blockDim.x + threadIdx.x;
  const long long stride = (long long)gridDim.x * blockDim.x;
  for (; i < nch; i += stride) {
    const int tok = (int)(i >> 7);       // HD/8 = 128 chunks per token
    const int off = ((int)i & 127) << 3;
    const long long erow = ((long long)eidx[tok] * CAP + slot[tok]) * HD;
    const v8bf e8 = *(const v8bf*)(eo + erow + off);
    const v8bf r8 = *(const v8bf*)(res + (long long)tok * HD + off);
    const float s0 = gv[tok] * coef0[tok];
    const float s1 = coef1[tok];
    v8bf o;
#pragma unroll
    for (int j = 0; j < 8; ++j)
      o[j] = (bf16_t)((float)e8[j] * s0 + (float)r8[j] * s1);
    *(v8bf*)(mix + (long long)tok * HD + off) = o;
  }
}

// ---------------------------------------------------------------------------
extern "C" void kernel_launch(void* const* d_in, const int* in_sizes, int n_in,
                              void* d_out, int out_size, void* d_ws, size_t ws_size,
                              hipStream_t stream) {
  const float* x      = (const float*)d_in[0];
  const float* W_in   = (const float*)d_in[1];
  const float* b_in   = (const float*)d_in[2];
  const float* W_gate = (const float*)d_in[3];
  const float* W1     = (const float*)d_in[4];
  const float* b1     = (const float*)d_in[5];
  const float* W2     = (const float*)d_in[6];
  const float* b2     = (const float*)d_in[7];
  const float* Wr1    = (const float*)d_in[8];
  const float* br1    = (const float*)d_in[9];
  const float* Wr2    = (const float*)d_in[10];
  const float* br2    = (const float*)d_in[11];
  const float* W_coef = (const float*)d_in[12];
  const float* b_coef = (const float*)d_in[13];
  const float* W_out  = (const float*)d_in[14];
  const float* b_out  = (const float*)d_in[15];
  float* out = (float*)d_out;
  (void)in_sizes; (void)n_in; (void)out_size; (void)ws_size;

  char* w = (char*)d_ws;
  auto alloc = [&](size_t bytes) -> void* {
    void* p = (void*)w;
    w += (bytes + 255) & ~(size_t)255;
    return p;
  };
  const long long HH = (long long)HD * HD;

  bf16_t* x_bf  = (bf16_t*)alloc((size_t)TOK * HD * 2);
  bf16_t* WinT  = (bf16_t*)alloc((size_t)HH * 2);
  bf16_t* W1T   = (bf16_t*)alloc((size_t)NE * HH * 2);
  bf16_t* W2T   = (bf16_t*)alloc((size_t)NE * HH * 2);
  bf16_t* Wr1T  = (bf16_t*)alloc((size_t)HH * 2);
  bf16_t* Wr2T  = (bf16_t*)alloc((size_t)HH * 2);
  bf16_t* WoutT = (bf16_t*)alloc((size_t)HH * 2);
  bf16_t* t_bf  = (bf16_t*)alloc((size_t)TOK * HD * 2);
  int*    eidx  = (int*)alloc((size_t)TOK * 4);
  int*    slot  = (int*)alloc((size_t)TOK * 4);
  int*    keep  = (int*)alloc((size_t)TOK * 4);
  float*  gmax  = (float*)alloc((size_t)TOK * 4);
  float*  gv    = (float*)alloc((size_t)TOK * 4);
  float*  coef0 = (float*)alloc((size_t)TOK * 4);
  float*  coef1 = (float*)alloc((size_t)TOK * 4);
  bf16_t* disp  = (bf16_t*)alloc((size_t)NE * CAP * HD * 2);
  bf16_t* hbuf  = (bf16_t*)alloc((size_t)NE * CAP * HD * 2);
  bf16_t* eo    = (bf16_t*)alloc((size_t)NE * CAP * HD * 2);
  bf16_t* rh    = (bf16_t*)alloc((size_t)TOK * HD * 2);
  bf16_t* res   = (bf16_t*)alloc((size_t)TOK * HD * 2);
  bf16_t* mixb  = (bf16_t*)alloc((size_t)TOK * HD * 2);

  // 1) precision conversion: x -> bf16; weights -> transposed bf16 [N][K]
  cvt_f32_to_bf16<<<2048, 256, 0, stream>>>(x, x_bf, (long long)TOK * HD);
  const dim3 tb(32, 8);
  transpose_cvt<<<dim3(HD / 32, HD / 32, 1),  tb, 0, stream>>>(W_in,  WinT,  HD, HD, 0,  0);
  transpose_cvt<<<dim3(HD / 32, HD / 32, NE), tb, 0, stream>>>(W1,    W1T,   HD, HD, HH, HH);
  transpose_cvt<<<dim3(HD / 32, HD / 32, NE), tb, 0, stream>>>(W2,    W2T,   HD, HD, HH, HH);
  transpose_cvt<<<dim3(HD / 32, HD / 32, 1),  tb, 0, stream>>>(Wr1,   Wr1T,  HD, HD, 0,  0);
  transpose_cvt<<<dim3(HD / 32, HD / 32, 1),  tb, 0, stream>>>(Wr2,   Wr2T,  HD, HD, 0,  0);
  transpose_cvt<<<dim3(HD / 32, HD / 32, 1),  tb, 0, stream>>>(W_out, WoutT, HD, HD, 0,  0);

  // 2) t = x @ W_in + b_in  -> bf16
  gemm_bf16<1><<<dim3(HD / BN, TOK / BM, 1), 256, 0, stream>>>(
      x_bf, WinT, b_in, nullptr, t_bf, HD, HD, 0, 0, 0, 0);

  // 3) gating + 2-way coef softmax; ordered capacity scan
  gate_coef<<<TOK, 128, 0, stream>>>(t_bf, W_gate, W_coef, b_coef,
                                     eidx, gmax, coef0, coef1);
  scan_assign<<<1, 32, 0, stream>>>(eidx, gmax, slot, keep, gv);

  // 4) dispatch tokens into [E, CAP, H]
  fill_zero_u4<<<2048, 256, 0, stream>>>((uint4*)disp,
                                         (long long)NE * CAP * HD / 8);
  dispatch_rows<<<TOK, 128, 0, stream>>>(t_bf, eidx, slot, keep, disp);

  // 5) expert FFN (batched over experts via blockIdx.z)
  gemm_bf16<2><<<dim3(HD / BN, CAP / BM, NE), 256, 0, stream>>>(
      disp, W1T, b1, nullptr, hbuf, HD, HD,
      (long long)CAP * HD, HH, HD, (long long)CAP * HD);
  gemm_bf16<1><<<dim3(HD / BN, CAP / BM, NE), 256, 0, stream>>>(
      hbuf, W2T, b2, nullptr, eo, HD, HD,
      (long long)CAP * HD, HH, HD, (long long)CAP * HD);

  // 6) residual expert on all tokens
  gemm_bf16<2><<<dim3(HD / BN, TOK / BM, 1), 256, 0, stream>>>(
      t_bf, Wr1T, br1, nullptr, rh, HD, HD, 0, 0, 0, 0);
  gemm_bf16<1><<<dim3(HD / BN, TOK / BM, 1), 256, 0, stream>>>(
      rh, Wr2T, br2, nullptr, res, HD, HD, 0, 0, 0, 0);

  // 7) gather + coefficient mix
  mix_kernel<<<2048, 256, 0, stream>>>(eo, res, eidx, slot, gv, coef0, coef1, mixb);

  // 8) out = mix @ W_out + b_out  -> f32
  gemm_bf16<0><<<dim3(HD / BN, TOK / BM, 1), 256, 0, stream>>>(
      mixb, WoutT, b_out, out, nullptr, HD, HD, 0, 0, 0, 0);
}
